// MaskHeadProposals_84722524881006
// MI455X (gfx1250) — compile-verified, matrix operand order checked by reading, not run on hardware
//
#include <hip/hip_runtime.h>
#include <hip/hip_bf16.h>

// Problem constants (match reference)
#define B_    8
#define NG_   64
#define NP_   2048
#define N_    (NG_ + NP_)   // 2112
#define M_    320
#define NSORT 4096          // next pow2 >= N_ for bitonic sort
#define NMS_THR 0.3f
#define THREADS 1024        // 32 wave32s per workgroup

typedef __attribute__((ext_vector_type(16))) _Float16 v16h;
typedef __attribute__((ext_vector_type(8)))  float    v8f;

// ---------------------------------------------------------------------------
// Kernel 1: pos_len[b] = sum(gt_classes[b, :]) for all 8 batches at once,
// via a single V_WMMA_F32_16X16X32_F16 row-sum against an all-ones B matrix.
//   A (16x32 f16): row r holds gt_classes[r/2, (r&1)*32 .. (r&1)*32+31].
//   Since B is all-ones, only the row->lane mapping matters (K-permutation
//   within a row is sum-invariant). Lane l (l<16) supplies elements 0..15 of
//   row l; lane l+16 supplies elements 16..31 of row l (disjoint, complete).
//   C[r][j] = row-sum of r for every j; col 0 lives in lane 0 (rows 0..7 in
//   c[0..7]) and lane 16 (rows 8..15 in c[0..7]).
// Launched with exactly 32 threads so EXEC is all-ones (WMMA requirement).
// ---------------------------------------------------------------------------
__global__ __launch_bounds__(32)
void poslen_wmma_kernel(const float* __restrict__ gt_classes,
                        int* __restrict__ pos_len) {
    const int lane = threadIdx.x & 31;
    const int r    = lane & 15;        // A-matrix row handled by this lane
    const int b    = r >> 1;           // batch index
    const int base = (r & 1) * 32 + ((lane >> 4) ? 16 : 0);

    v16h a;
#pragma unroll
    for (int e = 0; e < 16; ++e)
        a[e] = (_Float16)gt_classes[b * NG_ + base + e];

    v16h ones;
#pragma unroll
    for (int e = 0; e < 16; ++e)
        ones[e] = (_Float16)1.0f;

    v8f c = {};
    // 8 args: (neg_a, A, neg_b, B, c_mod, C, reuse_a, reuse_b)
    c = __builtin_amdgcn_wmma_f32_16x16x32_f16(
            false, a, false, ones, (short)0, c, false, false);

    // CDNA5 tensor-counter wait (TENSORcnt is 0 here; semantically a no-op).
    __builtin_amdgcn_s_wait_tensorcnt(0);

    if (lane == 0) {
#pragma unroll
        for (int bb = 0; bb < 4; ++bb)                 // rows 0..7 -> batches 0..3
            pos_len[bb] = (int)(c[2 * bb] + c[2 * bb + 1] + 0.5f);
    }
    if (lane == 16) {
#pragma unroll
        for (int bb = 0; bb < 4; ++bb)                 // rows 8..15 -> batches 4..7
            pos_len[4 + bb] = (int)(c[2 * bb] + c[2 * bb + 1] + 0.5f);
    }
}

// ---------------------------------------------------------------------------
// Kernel 2: one workgroup per batch. Entire batch state lives in LDS
// (~116 KB << 320 KB/WGP on CDNA5):
//   phase -1: GLOBAL_LOAD_ASYNC_TO_LDS_B128 streams all boxes (original
//             order) into LDS staging; the copy overlaps the bitonic sort
//             and is fenced with s_wait_asynccnt before first use.
//   phase 0: build u64 sort keys  (score_bits<<32 | (0xFFFFFFFF - idx))
//            -> descending sort == stable argsort(-scores) of the reference
//   phase 1: bitonic sort (4096 keys, 78 barrier steps)
//   phase 2: permute staged boxes into sorted order (LDS -> LDS)
//   phase 3: sequential NMS; each surviving head box is tested against all
//            later boxes by 1024 threads; suppressed heads skip the barrier
//   phase 4: scatter keep back to original index order, serial negative-rank
//            table (first pos_len kept negatives in original order)
//   phase 5: write out_cls / out_boxes / out_scores
// ---------------------------------------------------------------------------
__global__ __launch_bounds__(THREADS)
void nms_select_kernel(const float* __restrict__ cls_proposals,
                       const float* __restrict__ gt_classes,
                       const float* __restrict__ box_proposals,
                       const float* __restrict__ gt_boxes,
                       const float* __restrict__ proposal_scores,
                       const int*   __restrict__ pos_len_arr,
                       float*       __restrict__ out) {
    __shared__ unsigned long long skey[NSORT];      // 32768 B
    __shared__ float4             sbox[N_];         // 33792 B (sorted order)
    __shared__ float              sarea[N_];        //  8448 B
    __shared__ int                skeep[N_];        //  8448 B
    __shared__ float4             sbox_orig[N_];    // 33792 B (staging; dead
                                                    //   after phase 2)
    __shared__ int                neg_tab[M_];      //  1280 B
    // oneg (original-order neg flags, phase 4) aliases the staging buffer:
    // staging's last read is in phase 2, separated by phase-3 barriers.
    int* oneg = (int*)sbox_orig;

    const int b   = blockIdx.x;
    const int tid = threadIdx.x;

    const float* gcls = gt_classes      + (size_t)b * NG_;
    const float* pcls = cls_proposals   + (size_t)b * NP_;
    const float* gbox = gt_boxes        + (size_t)b * NG_ * 4;
    const float* pbox = box_proposals   + (size_t)b * NP_ * 4;
    const float* psc  = proposal_scores + (size_t)b * NP_;

    // CDNA5 prefetch path (global_prefetch_b8) for data the async engine
    // doesn't stage (cls / scores read during key build and selection).
    __builtin_prefetch(pcls, 0, 1);
    __builtin_prefetch(psc, 0, 1);

    // ---- Phase -1: async global->LDS staging of all boxes -----------------
    // ASYNCcnt-tracked copy; overlaps phases 0-1 entirely.
    for (int i = tid; i < N_; i += THREADS) {
        const float4* src = (i < NG_) ? (const float4*)gbox + i
                                      : (const float4*)pbox + (i - NG_);
        unsigned int dsaddr =
            (unsigned int)(size_t)&sbox_orig[i];       // LDS offset (addr[31:0])
        unsigned long long gaddr = (unsigned long long)(size_t)src;
        asm volatile("global_load_async_to_lds_b128 %0, %1, off"
                     :: "v"(dsaddr), "v"(gaddr)
                     : "memory");
    }

    // ---- Phase 0: keys ----------------------------------------------------
    for (int i = tid; i < NSORT; i += THREADS) {
        unsigned long long key = 0ull;   // pad: sorts after every real entry
        if (i < N_) {
            float score;
            if (i < NG_) {
                score = gcls[i];                       // sc_all head = gt_classes
            } else {
                int p = i - NG_;
                score = (pcls[p] > 0.0f) ? 0.0f : psc[p];
            }
            // scores >= 0 -> float bits are order-preserving as u32.
            unsigned int sb = __float_as_uint(score);
            key = ((unsigned long long)sb << 32) |
                  (unsigned long long)(0xFFFFFFFFu - (unsigned)i);
        }
        skey[i] = key;
    }
    __syncthreads();

    // ---- Phase 1: bitonic sort, descending --------------------------------
    for (int k = 2; k <= NSORT; k <<= 1) {
        for (int j = k >> 1; j > 0; j >>= 1) {
            for (int i = tid; i < NSORT; i += THREADS) {
                int l = i ^ j;
                if (l > i) {
                    unsigned long long ka = skey[i];
                    unsigned long long kb = skey[l];
                    bool desc = ((i & k) == 0);
                    if ((ka < kb) == desc) { skey[i] = kb; skey[l] = ka; }
                }
            }
            __syncthreads();
        }
    }

    // Fence the async staging copy (per-wave), then barrier so every wave
    // sees the fully staged boxes.
    asm volatile("s_wait_asynccnt 0x0" ::: "memory");
    __syncthreads();

    // ---- Phase 2: permute staged boxes into sorted order (LDS->LDS) -------
    for (int s = tid; s < N_; s += THREADS) {
        unsigned int orig =
            0xFFFFFFFFu - (unsigned int)(skey[s] & 0xFFFFFFFFull);
        float4 bx = sbox_orig[orig];
        sbox[s]  = bx;
        sarea[s] = (bx.z - bx.x) * (bx.w - bx.y);
        skeep[s] = 1;
    }
    __syncthreads();

    // ---- Phase 3: sequential NMS ------------------------------------------
    for (int i = 0; i < N_ - 1; ++i) {
        // Uniform branch: every thread reads the same LDS word (broadcast).
        if (skeep[i]) {
            const float4 bi = sbox[i];
            const float  ai = sarea[i];
            for (int s = i + 1 + tid; s < N_; s += THREADS) {
                if (skeep[s]) {
                    float4 bj = sbox[s];
                    float iw = fmaxf(fminf(bi.z, bj.z) - fmaxf(bi.x, bj.x), 0.0f);
                    float ih = fmaxf(fminf(bi.w, bj.w) - fmaxf(bi.y, bj.y), 0.0f);
                    float inter = iw * ih;
                    float iou = inter / (ai + sarea[s] - inter + 1e-9f);
                    if (iou > NMS_THR) skeep[s] = 0;
                }
            }
            __syncthreads();   // only pay the 32-wave barrier when work happened
        }
    }

    // ---- Phase 4a: scatter keep&neg back to original order ----------------
    // (oneg aliases sbox_orig; staging data is no longer needed.)
    for (int s = tid; s < N_; s += THREADS) {
        unsigned int orig =
            0xFFFFFFFFu - (unsigned int)(skey[s] & 0xFFFFFFFFull);
        float cls = (orig < NG_) ? gcls[orig] : pcls[orig - NG_];
        oneg[orig] = (skeep[s] != 0) && (cls == 0.0f);
    }
    for (int t = tid; t < M_; t += THREADS) neg_tab[t] = -1;
    __syncthreads();

    const int plen = pos_len_arr[b];

    // ---- Phase 4b: serial rank of negatives (first plen, original order) --
    if (tid == 0) {
        int r = 0;
        for (int i2 = 0; i2 < N_; ++i2) {
            if (oneg[i2]) {
                if (r < plen) neg_tab[r] = i2;
                ++r;
            }
        }
    }
    __syncthreads();

    // ---- Phase 5: outputs (cls | boxes | scores, concatenated flat) -------
    float* out_cls = out;                                       // [B][M]
    float* out_box = out + (size_t)B_ * M_;                     // [B][M][4]
    float* out_sc  = out + (size_t)B_ * M_ * 5;                 // [B][M]
    float* ocls_b  = out_cls + (size_t)b * M_;
    float* obox_b  = out_box + (size_t)b * M_ * 4;
    float* osc_b   = out_sc  + (size_t)b * M_;

    for (int j = tid; j < M_; j += THREADS) {
        int  idx   = 0;
        bool valid = false;
        if (j < plen) {
            idx = j; valid = true;
        } else {
            int np = neg_tab[j - plen];
            valid = (np >= 0);
            idx   = np;
        }
        float  ocls = 0.0f, osc = 0.0f;
        float4 obx  = make_float4(0.0f, 0.0f, 0.0f, 0.0f);
        if (valid) {
            if (idx < NG_) {
                ocls = gcls[idx];
                osc  = gcls[idx];                 // sc_all head = gt_classes
                obx  = ((const float4*)gbox)[idx];
            } else {
                int p = idx - NG_;
                ocls = pcls[p];
                osc  = (pcls[p] > 0.0f) ? 0.0f : psc[p];
                obx  = ((const float4*)pbox)[p];
            }
        }
        ocls_b[j]         = ocls;
        obox_b[4 * j + 0] = obx.x;
        obox_b[4 * j + 1] = obx.y;
        obox_b[4 * j + 2] = obx.z;
        obox_b[4 * j + 3] = obx.w;
        osc_b[j]          = osc;
    }
}

// ---------------------------------------------------------------------------
extern "C" void kernel_launch(void* const* d_in, const int* in_sizes, int n_in,
                              void* d_out, int out_size, void* d_ws, size_t ws_size,
                              hipStream_t stream) {
    const float* cls_proposals   = (const float*)d_in[0];  // (B, NP)
    const float* gt_classes      = (const float*)d_in[1];  // (B, NG)
    const float* box_proposals   = (const float*)d_in[2];  // (B, NP, 4)
    const float* gt_boxes        = (const float*)d_in[3];  // (B, NG, 4)
    const float* proposal_scores = (const float*)d_in[4];  // (B, NP)

    int* pos_len = (int*)d_ws;   // 8 ints of scratch, written every call

    poslen_wmma_kernel<<<1, 32, 0, stream>>>(gt_classes, pos_len);
    nms_select_kernel<<<B_, THREADS, 0, stream>>>(
        cls_proposals, gt_classes, box_proposals, gt_boxes,
        proposal_scores, pos_len, (float*)d_out);
}
